// SparseAttention_54511724920847
// MI455X (gfx1250) — matrix operand order repaired and checked by reading, not compiled, not run on hardware
//
#include <hip/hip_runtime.h>
#include <hip/hip_bf16.h>

// MI455X / gfx1250, wave32, WMMA f32_16x16x32_f16 + async LDS staging.

typedef __attribute__((ext_vector_type(16))) _Float16 v16h;
typedef __attribute__((ext_vector_type(8)))  _Float16 v8h;
typedef __attribute__((ext_vector_type(8)))  float    v8f;

#define HB_E   256
#define HB_H   8
#define HB_S   32
#define HB_B   8
#define HB_N   1024   // GS*GS
#define NVALID 4096

__device__ __forceinline__ v8f wmma_f16(v16h a, v16h b, v8f c) {
    return __builtin_amdgcn_wmma_f32_16x16x32_f16(
        /*neg_a=*/false, a, /*neg_b=*/false, b,
        /*c_mod=*/(short)0, c, /*reuse_a=*/false, /*reuse_b=*/false);
}

__device__ __forceinline__ v16h pack16(v8h lo, v8h hi) {
    v16h r;
#pragma unroll
    for (int j = 0; j < 8; ++j) { r[j] = lo[j]; r[8 + j] = hi[j]; }
    return r;
}

// Async global->LDS copy, 8 bytes per lane.  CDNA5 GLOBAL_LOAD_ASYNC_TO_LDS_B64
// (VGLOBAL opcode 97), tracked with ASYNCcnt.  ldsoff = workgroup-relative LDS
// byte address (low 32 bits of a generic pointer to __shared__).
__device__ __forceinline__ void async_ld_b64(unsigned ldsoff, const void* g) {
    asm volatile("global_load_async_to_lds_b64 %0, %1, off"
                 :: "v"(ldsoff), "v"(g) : "memory");
}
__device__ __forceinline__ void wait_async_le2() {
    asm volatile("s_wait_asynccnt 0x2" ::: "memory");
}
__device__ __forceinline__ void wait_async_0() {
    asm volatile("s_wait_asynccnt 0x0" ::: "memory");
}

// B-fragment of W^T for the projection GEMM: B[k=s][n=d] = W[h][d][s].
__device__ __forceinline__ v16h wfrag(const float* __restrict__ W, int h, int d, int half) {
    const float* p = W + ((size_t)(h * HB_S + d) * HB_S + 16 * half);
    v16h r;
#pragma unroll
    for (int j = 0; j < 16; ++j) r[j] = (_Float16)p[j];
    return r;
}

// ---------------------------------------------------------------------------
// Kernel 0: convert Wu (256x256 f32) -> f16
// ---------------------------------------------------------------------------
__global__ void cvt_wu_kernel(const float* __restrict__ Wu, _Float16* __restrict__ Wu16, int n) {
    int i = blockIdx.x * blockDim.x + threadIdx.x;
    if (i < n) Wu16[i] = (_Float16)Wu[i];
}

// ---------------------------------------------------------------------------
// Kernel 1: QKV projection.  block = 256 thr (8 waves), grid = H*B*8.
// Each wave: 16 rows of one (h,b); K-dim = S = 32 -> one WMMA k-step.
// Stores Q,K row-major f16 [hb][n][d] and V transposed f16 [hb][d][n].
// ---------------------------------------------------------------------------
__global__ void qkv_kernel(const float* __restrict__ x,
                           const float* __restrict__ Wq, const float* __restrict__ bq,
                           const float* __restrict__ Wk, const float* __restrict__ bk,
                           const float* __restrict__ Wv, const float* __restrict__ bv,
                           _Float16* __restrict__ Qp, _Float16* __restrict__ Kp,
                           _Float16* __restrict__ Vt) {
    int blk   = blockIdx.x;
    int hb    = blk >> 3;          // h*8 + b
    int h     = hb >> 3;
    int b     = hb & 7;
    int rowb  = blk & 7;
    int wave  = threadIdx.x >> 5;
    int lane  = threadIdx.x & 31;
    int half  = lane >> 4;
    int l15   = lane & 15;
    int n0    = rowb * 128 + wave * 16;

    // A-fragment of X: lane row n = n0 + l15; element j -> s = 16*half + j (contiguous).
    const float* xb = x + ((size_t)(b * HB_E + h * HB_S) * HB_N) + (n0 + l15);
    v16h a;
#pragma unroll
    for (int j = 0; j < 16; ++j) {
        int s = 16 * half + j;
        a[j] = (_Float16)xb[(size_t)s * HB_N];
    }

    v8f z = {};
    v8f q0 = wmma_f16(a, wfrag(Wq, h, l15, half), z);
    v8f q1 = wmma_f16(a, wfrag(Wq, h, 16 + l15, half), z);
    v8f k0 = wmma_f16(a, wfrag(Wk, h, l15, half), z);
    v8f k1 = wmma_f16(a, wfrag(Wk, h, 16 + l15, half), z);
    v8f v0 = wmma_f16(a, wfrag(Wv, h, l15, half), z);
    v8f v1 = wmma_f16(a, wfrag(Wv, h, 16 + l15, half), z);

    float bq0 = bq[h * HB_S + l15], bq1 = bq[h * HB_S + 16 + l15];
    float bk0 = bk[h * HB_S + l15], bk1 = bk[h * HB_S + 16 + l15];
    float bv0 = bv[h * HB_S + l15], bv1 = bv[h * HB_S + 16 + l15];

    _Float16* qd = Qp + (size_t)hb * HB_N * HB_S;
    _Float16* kd = Kp + (size_t)hb * HB_N * HB_S;
    _Float16* vd = Vt + (size_t)hb * HB_S * HB_N;

#pragma unroll
    for (int r = 0; r < 8; ++r) {
        int n = n0 + r + 8 * half;                       // C-layout row
        qd[(size_t)n * HB_S + l15]        = (_Float16)(q0[r] + bq0);
        qd[(size_t)n * HB_S + 16 + l15]   = (_Float16)(q1[r] + bq1);
        kd[(size_t)n * HB_S + l15]        = (_Float16)(k0[r] + bk0);
        kd[(size_t)n * HB_S + 16 + l15]   = (_Float16)(k1[r] + bk1);
        vd[(size_t)l15        * HB_N + n] = (_Float16)(v0[r] + bv0);
        vd[(size_t)(16 + l15) * HB_N + n] = (_Float16)(v1[r] + bv1);
    }
}

// ---------------------------------------------------------------------------
// Kernel 2: flash attention.  block = 256 thr (8 waves), grid = H*B*8.
// All 8 waves share one (h,b) and the same key chunks, so K/V chunks are
// staged once per block into LDS with double-buffered async copies
// (global_load_async_to_lds_b64 / s_wait_asynccnt), overlapping the next
// chunk's copy with the current chunk's WMMA + softmax.
// ---------------------------------------------------------------------------
__global__ void attn_kernel(const _Float16* __restrict__ Qp,
                            const _Float16* __restrict__ Kp,
                            const _Float16* __restrict__ Vt,
                            _Float16* __restrict__ concat) {
    __shared__ __align__(32) _Float16 Kch[2][32 * 32];  // [buf][key*32 + d]
    __shared__ __align__(32) _Float16 Vch[2][32 * 32];  // [buf][d*32 + key]
    __shared__ __align__(32) _Float16 plds[8][16 * 32]; // per-wave P staging

    int blk  = blockIdx.x;
    int hb   = blk >> 3;
    int h    = hb >> 3;
    int b    = hb & 7;
    int rowb = blk & 7;
    int tid  = threadIdx.x;
    int wave = tid >> 5;
    int lane = tid & 31;
    int half = lane >> 4;
    int l15  = lane & 15;
    int n0   = rowb * 128 + wave * 16;

    const _Float16* kbase = Kp + (size_t)hb * HB_N * HB_S;
    const _Float16* vbase = Vt + (size_t)hb * HB_S * HB_N;

    // LDS byte offsets for this thread's async-copy slices (8 B per lane).
    unsigned koff[2], voff[2];
    int vd8 = tid >> 3, vo8 = (tid & 7) * 8;            // V: d-row, byte within row
#pragma unroll
    for (int u = 0; u < 2; ++u) {
        koff[u] = (unsigned)(size_t)(&Kch[u][0]) + (unsigned)(tid * 8);
        voff[u] = (unsigned)(size_t)(&Vch[u][0]) + (unsigned)(vd8 * 64 + vo8);
    }

    // Issue chunk 0 copies into buffer 0.
    async_ld_b64(koff[0], (const char*)kbase + (size_t)tid * 8);
    async_ld_b64(voff[0], (const char*)vbase + (size_t)vd8 * 2048 + vo8);

    // Q A-fragment: lane row = n0 + l15; 16 consecutive d per lane.
    const _Float16* qrow = Qp + ((size_t)hb * HB_N + n0 + l15) * HB_S;
    v16h qa = *(const v16h*)(qrow + 16 * half);

    float m[8], lsum[8];
    v8f acc0 = {}, acc1 = {};
#pragma unroll
    for (int r = 0; r < 8; ++r) { m[r] = -1e30f; lsum[r] = 0.0f; }

    const float scale = 1.0f / 16.0f;   // 1/sqrt(E), E = 256

    for (int c = 0; c < 32; ++c) {
        int buf = c & 1;
        __syncthreads();  // everyone done reading buf^1 (previous chunk) before overwrite
        if (c + 1 < 32) {
            size_t knext = (size_t)(c + 1) * 2048;      // bytes: 32 keys * 32 d * 2B
            async_ld_b64(koff[buf ^ 1], (const char*)kbase + knext + (size_t)tid * 8);
            async_ld_b64(voff[buf ^ 1],
                         (const char*)vbase + (size_t)vd8 * 2048 + (size_t)(c + 1) * 64 + vo8);
            wait_async_le2();   // all but the 2 just-issued -> current buf landed
        } else {
            wait_async_0();
        }
        __syncthreads();  // current buf visible from every wave's copies

        // K^T B-fragments from LDS: lane col = key, 16 consecutive d (32B ds read).
        v16h kb0 = *(const v16h*)(&Kch[buf][l15 * 32 + 16 * half]);
        v16h kb1 = *(const v16h*)(&Kch[buf][(16 + l15) * 32 + 16 * half]);

        v8f z = {};
        v8f s0 = wmma_f16(qa, kb0, z);
        v8f s1 = wmma_f16(qa, kb1, z);

        // Online softmax (per-row state, rows = r + 8*half; 16-lane reductions).
#pragma unroll
        for (int r = 0; r < 8; ++r) {
            float sa = s0[r] * scale;
            float sb = s1[r] * scale;
            float t = fmaxf(sa, sb);
            t = fmaxf(t, __shfl_xor(t, 1, 32));
            t = fmaxf(t, __shfl_xor(t, 2, 32));
            t = fmaxf(t, __shfl_xor(t, 4, 32));
            t = fmaxf(t, __shfl_xor(t, 8, 32));
            float mn = fmaxf(m[r], t);
            float al = __expf(m[r] - mn);
            float pa = __expf(sa - mn);
            float pb = __expf(sb - mn);
            float ps = pa + pb;
            ps += __shfl_xor(ps, 1, 32);
            ps += __shfl_xor(ps, 2, 32);
            ps += __shfl_xor(ps, 4, 32);
            ps += __shfl_xor(ps, 8, 32);
            lsum[r] = lsum[r] * al + ps;
            m[r] = mn;
            acc0[r] *= al;
            acc1[r] *= al;
            plds[wave][(r + 8 * half) * 32 + l15]      = (_Float16)pa;
            plds[wave][(r + 8 * half) * 32 + 16 + l15] = (_Float16)pb;
        }
        __syncthreads();  // order P-tile LDS stores vs cross-lane fragment reads

        // P A-fragment: lane row = l15, 16 consecutive keys per lane.
        const _Float16* pr = &plds[wave][l15 * 32];
        v16h pfrag = *(const v16h*)(pr + 16 * half);

        // V B-fragments from LDS [d][key]: 16 consecutive keys per lane (32B).
        v16h vb0 = *(const v16h*)(&Vch[buf][l15 * 32 + 16 * half]);
        v16h vb1 = *(const v16h*)(&Vch[buf][(16 + l15) * 32 + 16 * half]);

        acc0 = wmma_f16(pfrag, vb0, acc0);
        acc1 = wmma_f16(pfrag, vb1, acc1);
    }

    // Epilogue: divide by row sums, store head-concatenated f16.
#pragma unroll
    for (int r = 0; r < 8; ++r) {
        int n = n0 + r + 8 * half;
        float inv = 1.0f / lsum[r];
        size_t base = ((size_t)(b * HB_N + n)) * HB_E + h * HB_S;
        concat[base + l15]      = (_Float16)(acc0[r] * inv);
        concat[base + 16 + l15] = (_Float16)(acc1[r] * inv);
    }
}

// ---------------------------------------------------------------------------
// Kernel 3: unify heads + masked-row gather.  out[i][:] = concat[mask[i]] @ Wu^T + bu.
// Wave tile = 16 gathered rows x 16 cols, K = 256 -> 8 WMMA k-steps.
// ---------------------------------------------------------------------------
__global__ void unify_kernel(const _Float16* __restrict__ concat,
                             const _Float16* __restrict__ Wu16,
                             const float* __restrict__ bu,
                             const int* __restrict__ mask,
                             float* __restrict__ out) {
    int w    = blockIdx.x * 8 + (threadIdx.x >> 5);  // 0..4095
    int mt   = w >> 4;                               // 0..255
    int nt   = w & 15;                               // 0..15
    int lane = threadIdx.x & 31;
    int half = lane >> 4;
    int l15  = lane & 15;

    int row = mask[mt * 16 + l15];                   // gather row in [0, 8192)
    const _Float16* arow = concat + (size_t)row * HB_E;
    const _Float16* brow = Wu16 + (size_t)(nt * 16 + l15) * HB_E;

    v8f acc = {};
#pragma unroll
    for (int ks = 0; ks < 8; ++ks) {
        v16h a = *(const v16h*)(arow + ks * 32 + 16 * half);
        v16h bf = *(const v16h*)(brow + ks * 32 + 16 * half);
        acc = wmma_f16(a, bf, acc);
    }

    float bias = bu[nt * 16 + l15];
#pragma unroll
    for (int r = 0; r < 8; ++r) {
        out[(size_t)(mt * 16 + r + 8 * half) * HB_E + nt * 16 + l15] = acc[r] + bias;
    }
}

// ---------------------------------------------------------------------------
extern "C" void kernel_launch(void* const* d_in, const int* in_sizes, int n_in,
                              void* d_out, int out_size, void* d_ws, size_t ws_size,
                              hipStream_t stream) {
    (void)in_sizes; (void)n_in; (void)out_size; (void)ws_size;

    const float* x    = (const float*)d_in[0];
    const int*   mask = (const int*)d_in[3];
    const float* Wq   = (const float*)d_in[4];
    const float* bq   = (const float*)d_in[5];
    const float* Wk   = (const float*)d_in[6];
    const float* bk   = (const float*)d_in[7];
    const float* Wv   = (const float*)d_in[8];
    const float* bv   = (const float*)d_in[9];
    const float* Wu   = (const float*)d_in[10];
    const float* bu   = (const float*)d_in[11];
    float*       out  = (float*)d_out;

    char* ws = (char*)d_ws;
    _Float16* Qp   = (_Float16*)(ws);                         // 4 MiB: [64][1024][32]
    _Float16* Kp   = (_Float16*)(ws + (4ull  << 20));         // 4 MiB
    _Float16* Vt   = (_Float16*)(ws + (8ull  << 20));         // 4 MiB: [64][32][1024]
    _Float16* cc   = (_Float16*)(ws + (12ull << 20));         // 4 MiB: [8192][256]
    _Float16* Wu16 = (_Float16*)(ws + (16ull << 20));         // 128 KiB

    cvt_wu_kernel<<<256, 256, 0, stream>>>(Wu, Wu16, HB_E * HB_E);
    qkv_kernel  <<<512, 256, 0, stream>>>(x, Wq, bq, Wk, bk, Wv, bv, Qp, Kp, Vt);
    attn_kernel <<<512, 256, 0, stream>>>(Qp, Kp, Vt, cc);
    unify_kernel<<<512, 256, 0, stream>>>(cc, Wu16, bu, mask, out);
}
